// AttnDecoderRNN_6184752906686
// MI455X (gfx1250) — compile-verified, matrix operand order checked by reading, not run on hardware
//
#include <hip/hip_runtime.h>
#include <math.h>

#define B_ 256
#define S_ 64
#define H_ 1024
#define V_ 32000
#define T_ 10

typedef __attribute__((ext_vector_type(8)))  __bf16 v8bf;
typedef __attribute__((ext_vector_type(16))) __bf16 v16bf;
typedef __attribute__((ext_vector_type(8)))  float  v8f;

__device__ __forceinline__ unsigned short f2bf(float x) {
  unsigned u = __float_as_uint(x);
  unsigned r = 0x7FFFu + ((u >> 16) & 1u);
  return (unsigned short)((u + r) >> 16);
}
__device__ __forceinline__ float sigmoid_(float x) { return 1.0f / (1.0f + expf(-x)); }

// ---------------------------------------------------------------------------
// bf16 NT GEMM: C[M,N] = A[M,K] * B[N,K]^T + bias[N]
// A, B row-major bf16 (K contiguous). One workgroup = 4 waves, tile 128x64.
// Each wave: 32x64 (2x4 grid of 16x16 WMMA tiles), v_wmma_f32_16x16x32_bf16.
//
// The 64x32 B tile is shared by all 4 waves -> staged through LDS with
// GLOBAL_LOAD_ASYNC_TO_LDS_B128 (ASYNCcnt), double-buffered, so the shared
// operand is fetched from L2 once per workgroup instead of once per wave.
// A fragments (unique per wave) stay as direct global b128 loads + prefetch.
//
// Fragment layout per CDNA5 ISA 7.12.2 (16-bit A/B):
//   lanes 0-15  : row/col = lane,    elems 0-7 = K k..k+7,  8-15 = K k+16..k+23
//   lanes 16-31 : row/col = lane-16, elems 0-7 = K k+8..15, 8-15 = K k+24..k+31
// ---------------------------------------------------------------------------
__global__ __launch_bounds__(128) void gemm_nt_bias_kernel(
    const unsigned short* __restrict__ A,
    const unsigned short* __restrict__ Bw,
    const float* __restrict__ bias,
    float* __restrict__ C,
    int K, int ldc)
{
  __shared__ unsigned short Bs[2][64 * 32];   // 2 x 4KB double buffer

  const int tid  = threadIdx.x;
  const int lane = tid & 31;
  const int wave = tid >> 5;
  const int m0 = blockIdx.y * 128 + wave * 32;
  const int n0 = blockIdx.x * 64;
  const int fr = lane & 15;          // fragment row (A) / column (B)
  const int ko = (lane >> 4) << 3;   // K sub-offset: 0 or 8

  const unsigned short* Ab = A + (size_t)m0 * K;
  const unsigned short* Bb = Bw + (size_t)n0 * K;

  // Async staging: the 4KB B tile is 256 16-byte chunks; thread `tid` moves
  // chunks tid and tid+128. Chunk c -> row c>>2, k-elems (c&3)*8..+7.
  const int c0row = tid >> 2;
  const int c0k   = (tid & 3) * 8;
  const int c1row = c0row + 32;                 // (tid+128)>>2
  const unsigned ldsBase = (unsigned)(size_t)(&Bs[0][0]);   // LDS byte offset
  const unsigned dst0 = (unsigned)(c0row * 32 + c0k) * 2;
  const unsigned dst1 = (unsigned)(c1row * 32 + c0k) * 2;
  const unsigned bufBytes = 64 * 32 * 2;

  v8f acc[2][4] = {};

  const int nsteps = K >> 5;

  // prologue: stage K-slice 0 into buffer 0
  {
    const unsigned short* g0 = Bb + (size_t)c0row * K + c0k;
    const unsigned short* g1 = Bb + (size_t)c1row * K + c0k;
    asm volatile("global_load_async_to_lds_b128 %0, %1, off\n\t"
                 "global_load_async_to_lds_b128 %2, %3, off"
                 :: "v"(ldsBase + dst0), "v"(g0),
                    "v"(ldsBase + dst1), "v"(g1)
                 : "memory");
  }

  for (int ks = 0; ks < nsteps; ++ks) {
    const int k = ks << 5;
    const int cur = ks & 1;

    // own async writes landed ...
    asm volatile("s_wait_asynccnt 0x0" ::: "memory");
    // ... and everyone's are visible
    __syncthreads();

    // overlap: stage the next K-slice into the other buffer
    if (ks + 1 < nsteps) {
      const unsigned short* g0 = Bb + (size_t)c0row * K + (k + 32) + c0k;
      const unsigned short* g1 = Bb + (size_t)c1row * K + (k + 32) + c0k;
      const unsigned d = ldsBase + (unsigned)(cur ^ 1) * bufBytes;
      asm volatile("global_load_async_to_lds_b128 %0, %1, off\n\t"
                   "global_load_async_to_lds_b128 %2, %3, off"
                   :: "v"(d + dst0), "v"(g0), "v"(d + dst1), "v"(g1)
                   : "memory");
    }

    // A fragments direct from global (per-wave unique rows)
    v16bf af[2];
#pragma unroll
    for (int i = 0; i < 2; ++i) {
      const unsigned short* p = Ab + (size_t)(i * 16 + fr) * K + k + ko;
      v8bf lo = *(const v8bf*)p;
      v8bf hi = *(const v8bf*)(p + 16);
      af[i] = __builtin_shufflevector(lo, hi, 0,1,2,3,4,5,6,7,8,9,10,11,12,13,14,15);
    }
    if (k + 32 < K)   // gfx1250 global_prefetch_b8 for next A slice
      __builtin_prefetch(Ab + (size_t)fr * K + k + 32 + ko, 0, 1);

    // B fragments from LDS (ds_load_b128)
    v16bf bfm[4];
    const unsigned short* Bl = &Bs[cur][0];
#pragma unroll
    for (int j = 0; j < 4; ++j) {
      const unsigned short* p = Bl + (j * 16 + fr) * 32 + ko;
      v8bf lo = *(const v8bf*)p;
      v8bf hi = *(const v8bf*)(p + 16);
      bfm[j] = __builtin_shufflevector(lo, hi, 0,1,2,3,4,5,6,7,8,9,10,11,12,13,14,15);
    }

#pragma unroll
    for (int i = 0; i < 2; ++i)
#pragma unroll
      for (int j = 0; j < 4; ++j)
        acc[i][j] = __builtin_amdgcn_wmma_f32_16x16x32_bf16(
            false, af[i], false, bfm[j], (short)0, acc[i][j], false, false);
  }

  // C/D layout: VGPR r -> M=r (lanes 0-15) / M=r+8 (lanes 16-31), N = lane&15
  const int ccol = lane & 15;
  const int crow = (lane >> 4) << 3;
#pragma unroll
  for (int i = 0; i < 2; ++i)
#pragma unroll
    for (int j = 0; j < 4; ++j) {
      const int col = n0 + j * 16 + ccol;
      const float bv = bias[col];
#pragma unroll
      for (int r = 0; r < 8; ++r) {
        const int row = m0 + i * 16 + crow + r;
        C[(size_t)row * ldc + col] = acc[i][j][r] + bv;
      }
    }
}

// ---------------------------------------------------------------------------
// Elementwise / helper kernels
// ---------------------------------------------------------------------------
__global__ void f32_to_bf16_kernel(const float* __restrict__ src,
                                   unsigned short* __restrict__ dst, size_t n) {
  size_t i = (size_t)blockIdx.x * 256 + threadIdx.x;
  size_t stride = (size_t)gridDim.x * 256;
  for (; i < n; i += stride) dst[i] = f2bf(src[i]);
}

__global__ void h_init_kernel(const float* __restrict__ eh,
                              float* __restrict__ h,
                              unsigned short* __restrict__ hb) {
  int i = blockIdx.x * 256 + threadIdx.x;   // B*H
  float v = eh[i];
  h[i] = v;
  hb[i] = f2bf(v);
}

// x[:, 0:H] = bf16(embedding[tok])
__global__ void embed_kernel(const int* __restrict__ target,
                             const float* __restrict__ emb,
                             unsigned short* __restrict__ x, int t) {
  int i = blockIdx.x * 256 + threadIdx.x;   // B*H
  int b = i >> 10, j = i & 1023;
  int tok = (t == 0) ? 0 : target[b * T_ + (t - 1)];
  x[(size_t)b * 2048 + j] = f2bf(emb[(size_t)tok * H_ + j]);
}

// scores[b,s] = sum_h tanh(wq[b,h] + Uk[b*S+s,h]) * Va[h] + Va_b
__global__ __launch_bounds__(256) void attn_score_kernel(
    const float* __restrict__ wq, const float* __restrict__ Uk,
    const float* __restrict__ Va, const float* __restrict__ Vab,
    float* __restrict__ scores) {
  int wave = threadIdx.x >> 5, lane = threadIdx.x & 31;
  int row = blockIdx.x * 8 + wave;          // 0 .. B*S-1
  int b = row >> 6;
  const float* uk = Uk + (size_t)row * H_;
  const float* wqb = wq + (size_t)b * H_;
  float acc = 0.f;
  for (int j = lane; j < H_; j += 32)
    acc += tanhf(wqb[j] + uk[j]) * Va[j];
#pragma unroll
  for (int m = 16; m > 0; m >>= 1) acc += __shfl_xor(acc, m, 32);
  if (lane == 0) scores[row] = acc + Vab[0];
}

// softmax over S=64 per batch row (one wave per row, 2 elems/lane);
// writes normalized weights to wbuf and to attentions output [b,t,s]
__global__ __launch_bounds__(256) void attn_softmax_kernel(
    const float* __restrict__ scores, float* __restrict__ wbuf,
    float* __restrict__ attn_out, int t) {
  int wave = threadIdx.x >> 5, lane = threadIdx.x & 31;
  int b = blockIdx.x * 8 + wave;
  float v0 = scores[b * S_ + lane];
  float v1 = scores[b * S_ + 32 + lane];
  float m = fmaxf(v0, v1);
#pragma unroll
  for (int o = 16; o > 0; o >>= 1) m = fmaxf(m, __shfl_xor(m, o, 32));
  float e0 = expf(v0 - m), e1 = expf(v1 - m);
  float s = e0 + e1;
#pragma unroll
  for (int o = 16; o > 0; o >>= 1) s += __shfl_xor(s, o, 32);
  float inv = 1.f / s;
  e0 *= inv; e1 *= inv;
  wbuf[b * S_ + lane] = e0;
  wbuf[b * S_ + 32 + lane] = e1;
  float* ao = attn_out + (size_t)b * (T_ * S_) + (size_t)t * S_;
  ao[lane] = e0;
  ao[lane + 32] = e1;
}

// x[:, H:2H] = bf16(sum_s w[b,s] * keys[b,s,h])
__global__ void ctx_kernel(const float* __restrict__ wbuf,
                           const float* __restrict__ keys,
                           unsigned short* __restrict__ x) {
  int i = blockIdx.x * 256 + threadIdx.x;   // B*H
  int b = i >> 10, j = i & 1023;
  const float* wb = wbuf + b * S_;
  const float* kb = keys + ((size_t)b * S_) * H_ + j;
  float acc = 0.f;
#pragma unroll 8
  for (int s = 0; s < S_; ++s) acc += wb[s] * kb[(size_t)s * H_];
  x[(size_t)b * 2048 + H_ + j] = f2bf(acc);
}

// GRU gate fusion: h <- (1-z)*n + z*h ; also refresh bf16 copy of h
__global__ void gru_kernel(const float* __restrict__ gi,
                           const float* __restrict__ gh,
                           float* __restrict__ h,
                           unsigned short* __restrict__ hb) {
  int i = blockIdx.x * 256 + threadIdx.x;   // B*H
  int b = i >> 10, j = i & 1023;
  const float* gib = gi + (size_t)b * 3072;
  const float* ghb = gh + (size_t)b * 3072;
  float r = sigmoid_(gib[j] + ghb[j]);
  float z = sigmoid_(gib[j + 1024] + ghb[j + 1024]);
  float n = tanhf(gib[j + 2048] + r * ghb[j + 2048]);
  float h2 = (1.f - z) * n + z * h[i];
  h[i] = h2;
  hb[i] = f2bf(h2);
}

// in-place log-softmax over V=32000, one 256-thread block per (b,t) row
__global__ __launch_bounds__(256) void logsoftmax_kernel(float* __restrict__ dec, int t) {
  __shared__ float red[256];
  int b = blockIdx.x, tid = threadIdx.x;
  float* row = dec + (size_t)b * (T_ * V_) + (size_t)t * V_;
  float m = -3.0e38f;
  for (int i = tid; i < V_; i += 256) m = fmaxf(m, row[i]);
  red[tid] = m; __syncthreads();
  for (int s = 128; s > 0; s >>= 1) {
    if (tid < s) red[tid] = fmaxf(red[tid], red[tid + s]);
    __syncthreads();
  }
  m = red[0]; __syncthreads();
  float sum = 0.f;
  for (int i = tid; i < V_; i += 256) sum += expf(row[i] - m);
  red[tid] = sum; __syncthreads();
  for (int s = 128; s > 0; s >>= 1) {
    if (tid < s) red[tid] += red[tid + s];
    __syncthreads();
  }
  float lse = m + logf(red[0]);
  for (int i = tid; i < V_; i += 256) row[i] = row[i] - lse;
}

__global__ void hlast_kernel(const float* __restrict__ h, float* __restrict__ out) {
  int i = blockIdx.x * 256 + threadIdx.x;
  out[i] = h[i];
}

// ---------------------------------------------------------------------------
extern "C" void kernel_launch(void* const* d_in, const int* in_sizes, int n_in,
                              void* d_out, int out_size, void* d_ws, size_t ws_size,
                              hipStream_t stream) {
  (void)in_sizes; (void)n_in; (void)out_size; (void)ws_size;

  const float* keys   = (const float*)d_in[0];   // [B,S,H]
  const float* eh     = (const float*)d_in[1];   // [1,B,H]
  const int*   target = (const int*)d_in[2];     // [B,T]
  const float* emb    = (const float*)d_in[3];   // [V,H]
  const float* Wa_w   = (const float*)d_in[4];
  const float* Wa_b   = (const float*)d_in[5];
  const float* Ua_w   = (const float*)d_in[6];
  const float* Ua_b   = (const float*)d_in[7];
  const float* Va_w   = (const float*)d_in[8];
  const float* Va_b   = (const float*)d_in[9];
  const float* w_ih   = (const float*)d_in[10];  // [3H,2H]
  const float* w_hh   = (const float*)d_in[11];  // [3H,H]
  const float* b_ih   = (const float*)d_in[12];
  const float* b_hh   = (const float*)d_in[13];
  const float* out_w  = (const float*)d_in[14];  // [V,H]
  const float* out_b  = (const float*)d_in[15];

  float* dec_out = (float*)d_out;                          // [B,T,V]
  float* h_last  = dec_out + (size_t)B_ * T_ * V_;         // [B,H]
  float* attn    = h_last + (size_t)B_ * H_;               // [B,T,S]

  // workspace bump allocator (256B aligned)
  size_t off = 0;
  char* wsb = (char*)d_ws;
  auto take = [&](size_t bytes) -> char* {
    char* p = wsb + off;
    off += (bytes + 255) & ~(size_t)255;
    return p;
  };
  unsigned short* keys_bf = (unsigned short*)take((size_t)B_ * S_ * H_ * 2);
  unsigned short* Wa_bf   = (unsigned short*)take((size_t)H_ * H_ * 2);
  unsigned short* Ua_bf   = (unsigned short*)take((size_t)H_ * H_ * 2);
  unsigned short* wih_bf  = (unsigned short*)take((size_t)3 * H_ * 2 * H_ * 2);
  unsigned short* whh_bf  = (unsigned short*)take((size_t)3 * H_ * H_ * 2);
  unsigned short* outw_bf = (unsigned short*)take((size_t)V_ * H_ * 2);
  float*          Uk      = (float*)take((size_t)B_ * S_ * H_ * 4);
  float*          h       = (float*)take((size_t)B_ * H_ * 4);
  unsigned short* hb      = (unsigned short*)take((size_t)B_ * H_ * 2);
  float*          wq      = (float*)take((size_t)B_ * H_ * 4);
  float*          scores  = (float*)take((size_t)B_ * S_ * 4);
  float*          wbuf    = (float*)take((size_t)B_ * S_ * 4);
  unsigned short* xbf     = (unsigned short*)take((size_t)B_ * 2 * H_ * 2);
  float*          gi      = (float*)take((size_t)B_ * 3 * H_ * 4);
  float*          gh      = (float*)take((size_t)B_ * 3 * H_ * 4);

  // --- one-time: fp32 -> bf16 weight/key conversion ---
  auto conv = [&](const float* s, unsigned short* d, size_t n) {
    f32_to_bf16_kernel<<<2048, 256, 0, stream>>>(s, d, n);
  };
  conv(keys,  keys_bf, (size_t)B_ * S_ * H_);
  conv(Wa_w,  Wa_bf,   (size_t)H_ * H_);
  conv(Ua_w,  Ua_bf,   (size_t)H_ * H_);
  conv(w_ih,  wih_bf,  (size_t)3 * H_ * 2 * H_);
  conv(w_hh,  whh_bf,  (size_t)3 * H_ * H_);
  conv(out_w, outw_bf, (size_t)V_ * H_);

  h_init_kernel<<<(B_ * H_) / 256, 256, 0, stream>>>(eh, h, hb);

  // --- one-time: Uk[b,s,k] = keys @ Ua_w^T + Ua_b  (M=B*S, N=H, K=H) ---
  gemm_nt_bias_kernel<<<dim3(H_ / 64, (B_ * S_) / 128), 128, 0, stream>>>(
      keys_bf, Ua_bf, Ua_b, Uk, H_, H_);

  // --- decode loop (10 steps, teacher forcing) ---
  for (int t = 0; t < T_; ++t) {
    // input embedding into x[:, 0:H]
    embed_kernel<<<(B_ * H_) / 256, 256, 0, stream>>>(target, emb, xbf, t);

    // wq = h @ Wa_w^T + Wa_b   (M=B, N=H, K=H)
    gemm_nt_bias_kernel<<<dim3(H_ / 64, B_ / 128), 128, 0, stream>>>(
        hb, Wa_bf, Wa_b, wq, H_, H_);

    // attention scores + softmax + attention output
    attn_score_kernel<<<(B_ * S_) / 8, 256, 0, stream>>>(wq, Uk, Va_w, Va_b, scores);
    attn_softmax_kernel<<<B_ / 8, 256, 0, stream>>>(scores, wbuf, attn, t);

    // context into x[:, H:2H]
    ctx_kernel<<<(B_ * H_) / 256, 256, 0, stream>>>(wbuf, keys, xbf);

    // gi = x @ w_ih^T + b_ih   (M=B, N=3H, K=2H)
    gemm_nt_bias_kernel<<<dim3((3 * H_) / 64, B_ / 128), 128, 0, stream>>>(
        xbf, wih_bf, b_ih, gi, 2 * H_, 3 * H_);
    // gh = h @ w_hh^T + b_hh   (M=B, N=3H, K=H)
    gemm_nt_bias_kernel<<<dim3((3 * H_) / 64, B_ / 128), 128, 0, stream>>>(
        hb, whh_bf, b_hh, gh, H_, 3 * H_);

    // GRU gates -> new h (fp32 + bf16)
    gru_kernel<<<(B_ * H_) / 256, 256, 0, stream>>>(gi, gh, h, hb);

    // logits for this step directly into dec_out[:, t, :] (ldc = T*V)
    gemm_nt_bias_kernel<<<dim3(V_ / 64, B_ / 128), 128, 0, stream>>>(
        hb, outw_bf, out_b, dec_out + (size_t)t * V_, H_, T_ * V_);

    // in-place log-softmax over V
    logsoftmax_kernel<<<B_, 256, 0, stream>>>(dec_out, t);
  }

  hlast_kernel<<<(B_ * H_) / 256, 256, 0, stream>>>(h, h_last);
}
